// SelfAttention_66357244723910
// MI455X (gfx1250) — matrix operand order repaired and checked by reading, not compiled, not run on hardware
//
#include <hip/hip_runtime.h>
#include <hip/hip_bf16.h>
#include <cstdint>
#include <cstddef>

// Problem sizes (fixed by the reference)
constexpr int BB = 16;
constexpr int SS = 2048;
constexpr int HH = 1024;

typedef __attribute__((ext_vector_type(16))) _Float16 v16h;
typedef __attribute__((ext_vector_type(8)))  _Float16 v8h;
typedef __attribute__((ext_vector_type(8)))  float    v8f;

// Pointer/vector types for the async global->LDS builtin (per hipcc diagnostic:
// param is 'int __vector(4) __device__ *', i.e. AS1 pointer to 16B int vector).
typedef int v4i __attribute__((vector_size(4 * sizeof(int))));
typedef __attribute__((address_space(1))) v4i* as1_v4i_p;
typedef __attribute__((address_space(3))) v4i* as3_v4i_p;

// LDS B-tile geometry: 128 rows (n) x 32 halves (k), padded row stride = 40 halves
// (40*2B = 80B row stride -> the 16 lanes of a fragment read hit 16 distinct banks)
constexpr int BROW   = 40;
constexpr int BTROWS = 128;

// ---------------------------------------------------------------------------
// CDNA5 async global->LDS copy (ASYNCcnt-tracked) with inline-asm fallback
// ---------------------------------------------------------------------------
__device__ __forceinline__ void async_b128_g2l(const void* g, void* l) {
#if __has_builtin(__builtin_amdgcn_global_load_async_to_lds_b128)
  __builtin_amdgcn_global_load_async_to_lds_b128(
      (as1_v4i_p)(void*)g, (as3_v4i_p)(void*)l, 0, 0);
#else
  unsigned lofs = (unsigned)(uintptr_t)l;   // generic->LDS: low 32 bits are LDS offset
  asm volatile("global_load_async_to_lds_b128 %0, %1, off"
               :: "v"(lofs), "v"(g) : "memory");
#endif
}

__device__ __forceinline__ void wait_async0() {
#if __has_builtin(__builtin_amdgcn_s_wait_asynccnt)
  __builtin_amdgcn_s_wait_asynccnt(0);
#else
  asm volatile("s_wait_asynccnt 0x0" ::: "memory");
#endif
}

// Stage one 128x32 f16 B tile (rows = columns of the GEMM, contiguous k) into LDS.
// 512 16B chunks; 256 threads issue 2 async b128 copies each.
__device__ __forceinline__ void stage_b_tile(const _Float16* __restrict__ src,
                                             _Float16* lds, int tid) {
#pragma unroll
  for (int h = 0; h < 2; ++h) {
    const int c   = tid + h * 256;
    const int row = c >> 2;
    const int ch  = c & 3;
    async_b128_g2l(src + (size_t)row * HH + ch * 8, lds + row * BROW + ch * 8);
  }
}

// Read a 32x16 B fragment (column n-block t, k-half per lane) from the LDS tile.
__device__ __forceinline__ v16h read_b_frag(const _Float16* lds, int t, int nl, int hs) {
  const _Float16* bp = lds + (t * 16 + nl) * BROW + hs * 8;
  v8h b0 = *(const v8h*)bp;
  v8h b1 = *(const v8h*)(bp + 16);
  v16h b;
#pragma unroll
  for (int i = 0; i < 8; ++i) { b[i] = b0[i]; b[8 + i] = b1[i]; }
  return b;
}

// ---------------------------------------------------------------------------
// Weight transpose + f32->f16 convert:  Wt[n*HH + k] = (f16) W[k*HH + n]
// ---------------------------------------------------------------------------
__global__ __launch_bounds__(256)
void wtrans_kernel(const float* __restrict__ W, _Float16* __restrict__ Wt) {
  int i = blockIdx.x * 256 + threadIdx.x;   // i = k*HH + n, reads coalesced
  int k = i >> 10;
  int n = i & (HH - 1);
  Wt[(size_t)n * HH + k] = (_Float16)W[i];
}

__global__ __launch_bounds__(256)
void zero_kernel(float* __restrict__ p, int n) {
  int i = blockIdx.x * 256 + threadIdx.x;
  if (i < n) p[i] = 0.0f;
}

// ---------------------------------------------------------------------------
// QKV projection: Out[m, n] = f16( sum_k x[m,k] * W[k,n] + bias[n] )
// A = x (f32, converted in-register), B = Wt staged via async->LDS double buffer.
// Block: 256 threads = 8 waves; wave computes 16(M) x 128(N). Grid (256, 8).
// ---------------------------------------------------------------------------
__global__ __launch_bounds__(256)
void qkv_kernel(const float* __restrict__ x, const _Float16* __restrict__ Wt,
                const float* __restrict__ bias, _Float16* __restrict__ Out) {
  __shared__ _Float16 ldsB[2][BTROWS * BROW];
  const int tid  = threadIdx.x;
  const int lane = tid & 31;
  const int wave = tid >> 5;
  const int hs   = lane >> 4;
  const int nl   = lane & 15;
  const int m0   = blockIdx.x * 128 + wave * 16;
  const int n0   = blockIdx.y * 128;

  v8f acc[8] = {};
  const float*    arow  = x  + (size_t)(m0 + nl) * HH;
  const _Float16* bbase = Wt + (size_t)n0 * HH;

  stage_b_tile(bbase, ldsB[0], tid);
  wait_async0();
  __syncthreads();

  for (int k0 = 0; k0 < HH; k0 += 32) {
    const int cur = (k0 >> 5) & 1;
    if (k0 + 32 < HH)
      stage_b_tile(bbase + k0 + 32, ldsB[cur ^ 1], tid);

    // A fragment: lane holds row m0+nl, halves k0+8*hs+[0..7] and +16..23
    v16h a;
    const float* ap = arow + k0 + hs * 8;
#pragma unroll
    for (int i = 0; i < 8; ++i) {
      a[i]     = (_Float16)ap[i];
      a[8 + i] = (_Float16)ap[16 + i];
    }
    // Software-pipelined B fragment reads: fetch t+1 before consuming t.
    v16h b = read_b_frag(ldsB[cur], 0, nl, hs);
#pragma unroll
    for (int t = 0; t < 8; ++t) {
      v16h bn = b;
      if (t < 7) bn = read_b_frag(ldsB[cur], t + 1, nl, hs);
      acc[t] = __builtin_amdgcn_wmma_f32_16x16x32_f16(
          false, a, false, b, (short)0, acc[t], false, false);
      b = bn;
    }
    wait_async0();
    __syncthreads();
  }

#pragma unroll
  for (int t = 0; t < 8; ++t) {
    const int n = n0 + t * 16 + nl;
    const float bv = bias[n];
#pragma unroll
    for (int r = 0; r < 8; ++r) {
      const int m = m0 + r + 8 * hs;   // C/D layout: lanes16-31 hold M = r+8
      Out[(size_t)m * HH + n] = (_Float16)(acc[t][r] + bv);
    }
  }
}

// ---------------------------------------------------------------------------
// Scores (one batch): Sc[q, s] = scale * sum_d Q[q,d] * K[s,d]
// A = Q rows (f16), B = K rows staged via async->LDS double buffer (== Kᵀ frag).
// Grid: (S/128, S/128) = (16, 16), 256 threads.
// ---------------------------------------------------------------------------
__global__ __launch_bounds__(256)
void scores_kernel(const _Float16* __restrict__ Q, const _Float16* __restrict__ Km,
                   float* __restrict__ Sc, float scale) {
  __shared__ _Float16 ldsB[2][BTROWS * BROW];
  const int tid  = threadIdx.x;
  const int lane = tid & 31;
  const int wave = tid >> 5;
  const int hs   = lane >> 4;
  const int nl   = lane & 15;
  const int m0   = blockIdx.x * 128 + wave * 16;
  const int n0   = blockIdx.y * 128;

  v8f acc[8] = {};
  const _Float16* qrow  = Q  + (size_t)(m0 + nl) * HH;
  const _Float16* bbase = Km + (size_t)n0 * HH;

  stage_b_tile(bbase, ldsB[0], tid);
  wait_async0();
  __syncthreads();

  for (int k0 = 0; k0 < HH; k0 += 32) {
    const int cur = (k0 >> 5) & 1;
    if (k0 + 32 < HH)
      stage_b_tile(bbase + k0 + 32, ldsB[cur ^ 1], tid);

    v16h a;
    v8h lo = *(const v8h*)(qrow + k0 + hs * 8);
    v8h hi = *(const v8h*)(qrow + k0 + hs * 8 + 16);
#pragma unroll
    for (int i = 0; i < 8; ++i) { a[i] = lo[i]; a[8 + i] = hi[i]; }

    v16h b = read_b_frag(ldsB[cur], 0, nl, hs);
#pragma unroll
    for (int t = 0; t < 8; ++t) {
      v16h bn = b;
      if (t < 7) bn = read_b_frag(ldsB[cur], t + 1, nl, hs);
      acc[t] = __builtin_amdgcn_wmma_f32_16x16x32_f16(
          false, a, false, b, (short)0, acc[t], false, false);
      b = bn;
    }
    wait_async0();
    __syncthreads();
  }

#pragma unroll
  for (int t = 0; t < 8; ++t) {
    const int n = n0 + t * 16 + nl;
#pragma unroll
    for (int r = 0; r < 8; ++r) {
      const int m = m0 + r + 8 * hs;
      Sc[(size_t)m * SS + n] = acc[t][r] * scale;
    }
  }
}

// ---------------------------------------------------------------------------
// Row softmax stats (one batch): rrow[s] = max_k Sc[s,k] + log(sum_k exp(..-max))
// One wave32 per row, online max/sum, shuffle tree combine. Grid: S/8 blocks.
// ---------------------------------------------------------------------------
__global__ __launch_bounds__(256)
void rowstat_kernel(const float* __restrict__ Sc, float* __restrict__ rrow) {
  const int lane = threadIdx.x & 31;
  const int row  = blockIdx.x * 8 + (threadIdx.x >> 5);
  const float* p = Sc + (size_t)row * SS;

  float m = -3.402823466e38f, z = 0.0f;
  for (int i = lane; i < SS; i += 32) {
    float v  = p[i];
    float mn = fmaxf(m, v);
    z = z * __expf(m - mn) + __expf(v - mn);
    m = mn;
  }
  for (int off = 16; off > 0; off >>= 1) {
    float m2 = __shfl_xor(m, off, 32);
    float z2 = __shfl_xor(z, off, 32);
    float mn = fmaxf(m, m2);
    z = z * __expf(m - mn) + z2 * __expf(m2 - mn);
    m = mn;
  }
  if (lane == 0) rrow[row] = m + __logf(z);
}

// ---------------------------------------------------------------------------
// Column accumulation (one batch): c[k] += (1/S) * sum_s exp(Sc[s,k] - rrow[s])
// Thread t owns columns t, t+256, ...; block owns a 256-row strip. Grid: S/256.
// ---------------------------------------------------------------------------
__global__ __launch_bounds__(256)
void caccum_kernel(const float* __restrict__ Sc, const float* __restrict__ rrow,
                   float* __restrict__ c, float invS) {
  const int t  = threadIdx.x;
  const int s0 = blockIdx.x * 256;

  float acc[8] = {};
  for (int s = s0; s < s0 + 256; ++s) {
    const float rs = rrow[s];
    const float* p = Sc + (size_t)s * SS;
#pragma unroll
    for (int j = 0; j < 8; ++j) acc[j] += __expf(p[t + j * 256] - rs);
  }
#pragma unroll
  for (int j = 0; j < 8; ++j)
    __hip_atomic_fetch_add(&c[t + j * 256], acc[j] * invS,
                           __ATOMIC_RELAXED, __HIP_MEMORY_SCOPE_AGENT);
}

// ---------------------------------------------------------------------------
// Output GEMV (one batch): out[d] = sum_k c[k] * V[k,d]. Grid: H/256.
// ---------------------------------------------------------------------------
__global__ __launch_bounds__(256)
void outgemv_kernel(const float* __restrict__ c, const _Float16* __restrict__ V,
                    float* __restrict__ out) {
  const int d = blockIdx.x * 256 + threadIdx.x;
  float acc = 0.0f;
  for (int k = 0; k < SS; ++k)
    acc += c[k] * (float)V[(size_t)k * HH + d];
  out[d] = acc;
}

// ---------------------------------------------------------------------------
extern "C" void kernel_launch(void* const* d_in, const int* in_sizes, int n_in,
                              void* d_out, int out_size, void* d_ws, size_t ws_size,
                              hipStream_t stream) {
  const float* x  = (const float*)d_in[0];
  const float* Wq = (const float*)d_in[1];
  const float* bq = (const float*)d_in[2];
  const float* Wk = (const float*)d_in[3];
  const float* bk = (const float*)d_in[4];
  const float* Wv = (const float*)d_in[5];
  const float* bv = (const float*)d_in[6];
  float* out = (float*)d_out;

  // Workspace layout (256B-aligned regions), ~215 MB total
  char* ws = (char*)d_ws;
  size_t off = 0;
  auto alloc = [&](size_t bytes) -> void* {
    void* p = ws + off;
    off = (off + bytes + 255) & ~(size_t)255;
    return p;
  };
  _Float16* Qf  = (_Float16*)alloc((size_t)BB * SS * HH * sizeof(_Float16));
  _Float16* Kf  = (_Float16*)alloc((size_t)BB * SS * HH * sizeof(_Float16));
  _Float16* Vf  = (_Float16*)alloc((size_t)BB * SS * HH * sizeof(_Float16));
  _Float16* Wqt = (_Float16*)alloc((size_t)HH * HH * sizeof(_Float16));
  _Float16* Wkt = (_Float16*)alloc((size_t)HH * HH * sizeof(_Float16));
  _Float16* Wvt = (_Float16*)alloc((size_t)HH * HH * sizeof(_Float16));
  float*    Sc  = (float*)alloc((size_t)SS * SS * sizeof(float));   // per-batch, reused
  float*    rr  = (float*)alloc((size_t)SS * sizeof(float));        // per-batch, reused
  float*    cv  = (float*)alloc((size_t)BB * SS * sizeof(float));

  // 1) weights -> f16 transposed
  wtrans_kernel<<<(HH * HH) / 256, 256, 0, stream>>>(Wq, Wqt);
  wtrans_kernel<<<(HH * HH) / 256, 256, 0, stream>>>(Wk, Wkt);
  wtrans_kernel<<<(HH * HH) / 256, 256, 0, stream>>>(Wv, Wvt);
  zero_kernel<<<(BB * SS) / 256, 256, 0, stream>>>(cv, BB * SS);

  // 2) QKV projections over all batches: M = B*S = 32768
  dim3 gproj((BB * SS) / 128, HH / 128);
  qkv_kernel<<<gproj, 256, 0, stream>>>(x, Wqt, bq, Qf);
  qkv_kernel<<<gproj, 256, 0, stream>>>(x, Wkt, bk, Kf);
  qkv_kernel<<<gproj, 256, 0, stream>>>(x, Wvt, bv, Vf);

  const float scale = 1.0f / 32.0f;   // 1/sqrt(1024)
  const float invS  = 1.0f / (float)SS;

  // 3) per-batch attention (stream-serialized; Sc/rr reused)
  for (int b = 0; b < BB; ++b) {
    const _Float16* Qb = Qf + (size_t)b * SS * HH;
    const _Float16* Kb = Kf + (size_t)b * SS * HH;
    const _Float16* Vb = Vf + (size_t)b * SS * HH;
    float* cb = cv + (size_t)b * SS;

    scores_kernel<<<dim3(SS / 128, SS / 128), 256, 0, stream>>>(Qb, Kb, Sc, scale);
    rowstat_kernel<<<SS / 8, 256, 0, stream>>>(Sc, rr);
    caccum_kernel<<<SS / 256, 256, 0, stream>>>(Sc, rr, cb, invS);
    outgemv_kernel<<<HH / 256, 256, 0, stream>>>(cb, Vb, out + (size_t)b * HH);
  }
}
